// Model_72576357368112
// MI455X (gfx1250) — matrix operand order repaired and checked by reading, not compiled, not run on hardware
//
#include <hip/hip_runtime.h>
#include <hip/hip_bf16.h>
#include <math.h>

// ---------------- problem constants ----------------
#define BB 128
#define LL 128
#define DD 512
#define HH 8
#define HDIM 64
#define DFF 512
#define NLAYER 2
#define DN 128
#define NHH 16
#define NREL 16
#define NODES 16
#define KTOK 4
#define NN (BB*NODES)      // 2048 nodes
#define MM (BB*16)         // 2048 edges
#define ROWS (BB*LL)       // 16384 token rows

typedef __attribute__((ext_vector_type(16))) _Float16 v16h;
typedef __attribute__((ext_vector_type(8)))  _Float16 v8h;
typedef __attribute__((ext_vector_type(8)))  float    v8f;

#define DEV static __device__ __forceinline__

DEV v16h frag_combine(v8h lo, v8h hi) {
  return __builtin_shufflevector(lo, hi, 0,1,2,3,4,5,6,7,8,9,10,11,12,13,14,15);
}

// A-matrix fragment (16x32 f16) from a per-lane row pointer.
// halfword h -> K = kbase + h + 16*sel   (two b128 loads)
DEV v16h frag_a_ptr(const _Float16* row, int kbase, int sel) {
  const _Float16* p = row + kbase + 16*sel;
  v8h lo = *(const v8h*)(p);        // K = kbase+16*sel   .. +7
  v8h hi = *(const v8h*)(p + 8);    // K = kbase+16*sel+8 .. +15
  return frag_combine(lo, hi);
}

// B-matrix fragment where B = W^T from a per-lane row pointer of W (N x K row-major).
// halfword h -> K = kbase + h + 16*sel  (one contiguous 16-halfword read)
DEV v16h frag_b_ptr(const _Float16* row, int kbase, int sel) {
  const _Float16* p = row + kbase + 16*sel;
  v8h lo = *(const v8h*)(p);
  v8h hi = *(const v8h*)(p + 8);
  return frag_combine(lo, hi);
}

// B-matrix fragment from a row-major (K x N) source (strided per-element reads).
DEV v16h frag_b_col(const _Float16* mkn, int ld, int n, int kbase, int sel) {
  v16h r;
#pragma unroll
  for (int h = 0; h < 16; ++h) {
    int k = kbase + h + 16*sel;
    r[h] = mkn[(size_t)k * ld + n];
  }
  return r;
}

DEV v8f wmma_f16(v16h a, v16h b, v8f c) {
  return __builtin_amdgcn_wmma_f32_16x16x32_f16(false, a, false, b, (short)0, c, false, false);
}

// ---------------- f32 -> f16 convert ----------------
__global__ void f2h_kernel(const float* __restrict__ in, _Float16* __restrict__ out, int n) {
  int i = blockIdx.x * blockDim.x + threadIdx.x;
  if (i < n) out[i] = (_Float16)in[i];
}

// ---------------- embedding + positional encoding ----------------
__global__ void embed_kernel(const int* __restrict__ seq, const int* __restrict__ isvar,
                             const int* __restrict__ iscon, const float* __restrict__ emb,
                             const float* __restrict__ conceptv, const float* __restrict__ variablev,
                             float* __restrict__ xf, _Float16* __restrict__ xh) {
  int i = blockIdx.x * blockDim.x + threadIdx.x;
  if (i >= ROWS * DD) return;
  int d  = i & (DD - 1);
  int bl = i >> 9;               // / DD
  int l  = bl & (LL - 1);
  float e = emb[(size_t)seq[bl] * DD + d] * 22.627416997969522f;  // sqrt(512)
  float add;
  if (isvar[bl]) {
    add = variablev[d];
  } else if (iscon[bl]) {
    add = conceptv[d];
  } else {
    int de = d & ~1;
    float div = __expf(-(float)de * (9.210340371976184f / (float)DD)); // ln(1e4)
    float ang = (float)l * div;
    add = (d & 1) ? __cosf(ang) : __sinf(ang);
  }
  float v = e + add;
  xf[i] = v;
  xh[i] = (_Float16)v;
}

// ---------------- WMMA GEMM: Y = X(MxK,f16) * W(NxK,f16)^T + bias ----------------
// One wave computes a 16x64 strip: 4 accumulators reuse each A fragment 4x, and
// fragment loads are software-pipelined one K-step ahead of the WMMAs.
// grid (M/16, N/64), block 32.
__global__ void gemm_wT_kernel(const _Float16* __restrict__ X, const _Float16* __restrict__ W,
                               const float* __restrict__ bias,
                               float* __restrict__ Yf, _Float16* __restrict__ Yh,
                               int Mdim, int Ndim, int Kdim, int relu) {
  int tm = blockIdx.x;
  int tn0 = blockIdx.y * 4;
  int lane = threadIdx.x & 31;
  int sel = lane >> 4, lr = lane & 15;

  const _Float16* xrow = X + (size_t)(tm*16 + lr) * Kdim;
  const _Float16* wrow[4];
#pragma unroll
  for (int t = 0; t < 4; ++t) wrow[t] = W + (size_t)((tn0 + t)*16 + lr) * Kdim;

  v8f acc[4] = {};
  v16h a = frag_a_ptr(xrow, 0, sel);
  v16h b[4];
#pragma unroll
  for (int t = 0; t < 4; ++t) b[t] = frag_b_ptr(wrow[t], 0, sel);

  int k = 0;
  for (; k + 32 < Kdim; k += 32) {
    // issue next K-step's fragment loads before this step's WMMAs
    v16h an = frag_a_ptr(xrow, k + 32, sel);
    v16h bn[4];
#pragma unroll
    for (int t = 0; t < 4; ++t) bn[t] = frag_b_ptr(wrow[t], k + 32, sel);
    if (k + 64 < Kdim) {
      __builtin_prefetch(xrow + k + 64, 0, 1);
      __builtin_prefetch(wrow[0] + k + 64, 0, 1);
    }
#pragma unroll
    for (int t = 0; t < 4; ++t) acc[t] = wmma_f16(a, b[t], acc[t]);
    a = an;
#pragma unroll
    for (int t = 0; t < 4; ++t) b[t] = bn[t];
  }
#pragma unroll
  for (int t = 0; t < 4; ++t) acc[t] = wmma_f16(a, b[t], acc[t]);

#pragma unroll
  for (int t = 0; t < 4; ++t) {
    int n = (tn0 + t)*16 + lr;
    float bv = bias ? bias[n] : 0.f;
#pragma unroll
    for (int v = 0; v < 8; ++v) {
      int m = tm*16 + v + 8*sel;
      float val = acc[t][v] + bv;
      if (relu) val = fmaxf(val, 0.f);
      if (Yf) Yf[(size_t)m * Ndim + n] = val;
      if (Yh) Yh[(size_t)m * Ndim + n] = (_Float16)val;
    }
  }
}

// ---------------- fused attention: one block per (b,h), 8 waves ----------------
__global__ void attention_kernel(const _Float16* __restrict__ qkv, _Float16* __restrict__ out) {
  int bh = blockIdx.x;
  int b = bh / HH, h = bh % HH;
  const _Float16* base = qkv + (size_t)b * LL * (3*DD);
  const _Float16* Qp = base + h * HDIM;
  const _Float16* Kp = base + DD + h * HDIM;
  const _Float16* Vp = base + 2*DD + h * HDIM;
  int wave = threadIdx.x >> 5;
  int lane = threadIdx.x & 31;
  int sel = lane >> 4, lr = lane & 15;
  int m0 = wave * 16;

  __shared__ _Float16 Pl[LL * LL];     // 32 KB: probabilities (f16)
  __shared__ _Float16 Vl[LL * HDIM];   // 16 KB: V tile staged via async DMA

  // --- stage V into LDS with gfx1250 async-to-LDS (overlaps the S = QK^T WMMAs) ---
  {
    int tid = threadIdx.x;
#pragma unroll
    for (int c = 0; c < 4; ++c) {
      int chunk = tid + c*256;              // 0..1023 b128 transfers
      int r = chunk >> 3;                   // V row (key index) 0..127
      int col0 = (chunk & 7) * 8;           // V col base
      const _Float16* g = Vp + (size_t)r * (3*DD) + col0;
      unsigned lds = (unsigned)(size_t)(&Vl[0]) + (unsigned)(r*HDIM + col0) * 2u;
      asm volatile("global_load_async_to_lds_b128 %0, %1, off"
                   :: "v"(lds), "v"(g) : "memory");
    }
  }

  // --- S = Q K^T / sqrt(hd): hoisted Q fragments, 8 column tiles ---
  const _Float16* qrow = Qp + (size_t)(m0 + lr) * (3*DD);
  v16h aq0 = frag_a_ptr(qrow, 0, sel);
  v16h aq1 = frag_a_ptr(qrow, 32, sel);
  v8f s[8] = {};
#pragma unroll
  for (int nt = 0; nt < 8; ++nt) {
    const _Float16* krow = Kp + (size_t)(nt*16 + lr) * (3*DD);
    s[nt] = wmma_f16(aq0, frag_b_ptr(krow, 0, sel), s[nt]);
    s[nt] = wmma_f16(aq1, frag_b_ptr(krow, 32, sel), s[nt]);
  }
  const float scale = 0.125f;   // 1/sqrt(64)
#pragma unroll
  for (int nt = 0; nt < 8; ++nt)
#pragma unroll
    for (int v = 0; v < 8; ++v) s[nt][v] *= scale;

  // --- row-wise softmax (rows of a tile live across a 16-lane half) ---
  float rmax[8], rsum[8];
#pragma unroll
  for (int v = 0; v < 8; ++v) {
    float mx = -1e30f;
#pragma unroll
    for (int nt = 0; nt < 8; ++nt) mx = fmaxf(mx, s[nt][v]);
#pragma unroll
    for (int msk = 1; msk < 16; msk <<= 1) mx = fmaxf(mx, __shfl_xor(mx, msk, 32));
    rmax[v] = mx;
  }
#pragma unroll
  for (int v = 0; v < 8; ++v) {
    float sm = 0.f;
#pragma unroll
    for (int nt = 0; nt < 8; ++nt) {
      float e = __expf(s[nt][v] - rmax[v]);
      s[nt][v] = e;
      sm += e;
    }
#pragma unroll
    for (int msk = 1; msk < 16; msk <<= 1) sm += __shfl_xor(sm, msk, 32);
    rsum[v] = sm;
  }
  // stage P (f16) into LDS row-major for A-fragment reloads
#pragma unroll
  for (int nt = 0; nt < 8; ++nt)
#pragma unroll
    for (int v = 0; v < 8; ++v) {
      float p = s[nt][v] / rsum[v];
      Pl[(size_t)(m0 + v + 8*sel) * LL + nt*16 + lr] = (_Float16)p;
    }
  asm volatile("s_wait_asynccnt 0" ::: "memory");   // V DMA complete
  __syncthreads();

  // --- O = P V : 4 output tiles over hd=64, K-dim = 128 (P, V both from LDS) ---
  const _Float16* prow = &Pl[(size_t)(m0 + lr) * LL];
  v8f o[4] = {};
#pragma unroll
  for (int k = 0; k < LL; k += 32) {
    v16h a = frag_a_ptr(prow, k, sel);
#pragma unroll
    for (int nt = 0; nt < 4; ++nt) {
      v16h bfr = frag_b_col(&Vl[0], HDIM, nt*16 + lr, k, sel);
      o[nt] = wmma_f16(a, bfr, o[nt]);
    }
  }
#pragma unroll
  for (int nt = 0; nt < 4; ++nt)
#pragma unroll
    for (int v = 0; v < 8; ++v) {
      int m = m0 + v + 8*sel;
      out[((size_t)b * LL + m) * DD + h*HDIM + nt*16 + lr] = (_Float16)o[nt][v];
    }
}

// ---------------- residual add + layernorm (one block per row) ----------------
__global__ void ln_add_kernel(const float* __restrict__ x, const float* __restrict__ res,
                              const float* __restrict__ g, const float* __restrict__ bb,
                              float* __restrict__ xo, _Float16* __restrict__ xh) {
  int row = blockIdx.x;
  int t = threadIdx.x;      // 256 threads, 2 elements each
  const float* xr = x + (size_t)row * DD;
  const float* rr = res + (size_t)row * DD;
  float a0 = xr[t] + rr[t];
  float a1 = xr[t + 256] + rr[t + 256];
  __shared__ float red[256];
  red[t] = a0 + a1;
  __syncthreads();
  for (int off = 128; off > 0; off >>= 1) { if (t < off) red[t] += red[t + off]; __syncthreads(); }
  float mu = red[0] * (1.f / (float)DD);
  __syncthreads();
  float d0 = a0 - mu, d1 = a1 - mu;
  red[t] = d0*d0 + d1*d1;
  __syncthreads();
  for (int off = 128; off > 0; off >>= 1) { if (t < off) red[t] += red[t + off]; __syncthreads(); }
  float rstd = rsqrtf(red[0] * (1.f / (float)DD) + 1e-5f);
  float o0 = d0 * rstd * g[t]       + bb[t];
  float o1 = d1 * rstd * g[t + 256] + bb[t + 256];
  xo[(size_t)row*DD + t]       = o0;
  xo[(size_t)row*DD + t + 256] = o1;
  xh[(size_t)row*DD + t]       = (_Float16)o0;
  xh[(size_t)row*DD + t + 256] = (_Float16)o1;
}

// ---------------- gather + segment-sum to nodes ----------------
__global__ void node_kernel(const float* __restrict__ enc, const int* __restrict__ idx,
                            float* __restrict__ hf, _Float16* __restrict__ hh) {
  int j = blockIdx.x;       // node 0..NN-1
  int d = threadIdx.x;      // 0..DN-1
  int b = j >> 4;           // NODES=16 nodes per batch
  float s = 0.f;
#pragma unroll
  for (int t = 0; t < KTOK; ++t) {
    int tok = idx[j*KTOK + t];
    s += enc[((size_t)b * LL + tok) * DN + d];
  }
  hf[(size_t)j*DN + d] = s;
  hh[(size_t)j*DN + d] = (_Float16)s;
}

// ---------------- init logits / counters ----------------
__global__ void zero_kernel(float* p, int n) {
  int i = blockIdx.x * blockDim.x + threadIdx.x;
  if (i < n) p[i] = 0.f;
}
__global__ void init_misc_kernel(int* cnt, int* flags) {
  int t = threadIdx.x;
  if (t < BB) flags[t] = 1;
  if (t == 0) *cnt = 0;
}

// ---------------- NTL bilinear scorer: one wave per (16-row tile, (r,h)) ----------------
__global__ void ntl_kernel(const _Float16* __restrict__ hh, const float* __restrict__ hf,
                           const int* __restrict__ srcv, const int* __restrict__ dstv,
                           const _Float16* __restrict__ wh, const float* __restrict__ vv,
                           const float* __restrict__ bp, const float* __restrict__ uu,
                           float* __restrict__ logit) {
  int mt = blockIdx.x;           // 0..MM/16-1
  int rh = blockIdx.y;           // 0..NREL*NHH-1
  int lane = threadIdx.x & 31;
  int sel = lane >> 4, lr = lane & 15;
  const _Float16* wrh = wh + (size_t)rh * DN * DN;

  int drow = dstv[mt*16 + lr];
  const _Float16* ap = hh + (size_t)drow * DN;
  int srows[8];
#pragma unroll
  for (int v = 0; v < 8; ++v) srows[v] = srcv[mt*16 + v + 8*sel];

  // hoist the gathered A (tl) fragments once; reuse across all 8 N-tiles
  v16h afr[4];
#pragma unroll
  for (int kk = 0; kk < 4; ++kk) afr[kk] = frag_a_ptr(ap, kk*32, sel);

  // bil[m] = sum_i hd[m,i] * (TL W^T)[m,i]
  v8f bil = {};
#pragma unroll
  for (int nt = 0; nt < 8; ++nt) {
    const _Float16* wr = wrh + (size_t)(nt*16 + lr) * DN;
    v8f y = {};
#pragma unroll
    for (int kk = 0; kk < 4; ++kk) y = wmma_f16(afr[kk], frag_b_ptr(wr, kk*32, sel), y);
#pragma unroll
    for (int v = 0; v < 8; ++v) {
      float hdv = hf[(size_t)srows[v] * DN + nt*16 + lr];
      bil[v] += hdv * y[v];
    }
  }
#pragma unroll
  for (int v = 0; v < 8; ++v) {
#pragma unroll
    for (int msk = 1; msk < 16; msk <<= 1) bil[v] += __shfl_xor(bil[v], msk, 32);
  }
  __shared__ float bilbuf[16];
  if (lr == 0) {
#pragma unroll
    for (int v = 0; v < 8; ++v) bilbuf[v + 8*sel] = bil[v];
  }
  __syncthreads();

  if (threadIdx.x < 16) {
    int m = mt*16 + (int)threadIdx.x;
    int si = srcv[m], di = dstv[m];
    const float* vp = vv + (size_t)rh * (2*DN);
    float lin = 0.f;
    for (int k = 0; k < DN; ++k) lin += vp[k]      * hf[(size_t)si*DN + k];
    for (int k = 0; k < DN; ++k) lin += vp[DN + k] * hf[(size_t)di*DN + k];
    float z = tanhf(bilbuf[threadIdx.x] + lin + bp[rh]);
    int r = rh >> 4;    // rh = r*NHH + h
    atomicAdd(&logit[(size_t)m * NREL + r], uu[rh] * z);
  }
}

// ---------------- argmax / accuracy / exact-match ----------------
__global__ void score_kernel(const float* __restrict__ logit, const int* __restrict__ rel,
                             int* cnt, int* flags) {
  int m = blockIdx.x * blockDim.x + threadIdx.x;
  if (m >= MM) return;
  const float* lp = logit + (size_t)m * NREL;
  int best = 0; float bv = lp[0];
#pragma unroll
  for (int r = 1; r < NREL; ++r) { float v = lp[r]; if (v > bv) { bv = v; best = r; } }
  int eq = (rel[m] == best) ? 1 : 0;
  atomicAdd(cnt, eq);
  if (!eq) atomicExch(&flags[m >> 4], 0);   // EDGES=16 per batch
}
__global__ void final_kernel(const int* cnt, const int* flags, float* out) {
  if (threadIdx.x == 0) {
    int s = 0;
    for (int i = 0; i < BB; ++i) s += flags[i];
    out[MM * NREL]     = (float)(*cnt) / (float)MM;
    out[MM * NREL + 1] = (float)s / (float)BB;
  }
}

// ---------------- host launcher ----------------
static inline char* ws_take(void* base, size_t& off, size_t bytes) {
  char* p = (char*)base + off;
  off += (bytes + 255) & ~(size_t)255;
  return p;
}

extern "C" void kernel_launch(void* const* d_in, const int* in_sizes, int n_in,
                              void* d_out, int out_size, void* d_ws, size_t ws_size,
                              hipStream_t stream) {
  (void)in_sizes; (void)n_in; (void)out_size; (void)ws_size;
  const int*   seq       = (const int*)d_in[0];
  const int*   isvar     = (const int*)d_in[1];
  const int*   iscon     = (const int*)d_in[2];
  const int*   idx       = (const int*)d_in[5];
  const int*   srcv      = (const int*)d_in[7];
  const int*   dstv      = (const int*)d_in[8];
  const int*   rel       = (const int*)d_in[9];
  const float* emb       = (const float*)d_in[10];
  const float* conceptv  = (const float*)d_in[11];
  const float* variablev = (const float*)d_in[12];
  const float* in_w      = (const float*)d_in[13];
  const float* in_b      = (const float*)d_in[14];
  const float* out_w     = (const float*)d_in[15];
  const float* out_b     = (const float*)d_in[16];
  const float* ln1_g     = (const float*)d_in[17];
  const float* ln1_b     = (const float*)d_in[18];
  const float* ln2_g     = (const float*)d_in[19];
  const float* ln2_b     = (const float*)d_in[20];
  const float* ff_w1     = (const float*)d_in[21];
  const float* ff_b1     = (const float*)d_in[22];
  const float* ff_w2     = (const float*)d_in[23];
  const float* ff_b2     = (const float*)d_in[24];
  const float* lin_w     = (const float*)d_in[25];
  const float* lin_b     = (const float*)d_in[26];
  const float* ntl_w     = (const float*)d_in[27];
  const float* ntl_v     = (const float*)d_in[28];
  const float* ntl_bp    = (const float*)d_in[29];
  const float* ntl_u     = (const float*)d_in[30];
  float* out = (float*)d_out;

  // ---- workspace carve ----
  size_t off = 0;
  float*    xf    = (float*)   ws_take(d_ws, off, (size_t)ROWS*DD*4);
  float*    t1f   = (float*)   ws_take(d_ws, off, (size_t)ROWS*DD*4);
  float*    encf  = (float*)   ws_take(d_ws, off, (size_t)ROWS*DN*4);
  float*    hf    = (float*)   ws_take(d_ws, off, (size_t)NN*DN*4);
  _Float16* xh    = (_Float16*)ws_take(d_ws, off, (size_t)ROWS*DD*2);
  _Float16* qkvh  = (_Float16*)ws_take(d_ws, off, (size_t)ROWS*3*DD*2);
  _Float16* attnh = (_Float16*)ws_take(d_ws, off, (size_t)ROWS*DD*2);
  _Float16* ffh   = (_Float16*)ws_take(d_ws, off, (size_t)ROWS*DFF*2);
  _Float16* hh    = (_Float16*)ws_take(d_ws, off, (size_t)NN*DN*2);
  _Float16* in_wh = (_Float16*)ws_take(d_ws, off, (size_t)NLAYER*3*DD*DD*2);
  _Float16* out_wh= (_Float16*)ws_take(d_ws, off, (size_t)NLAYER*DD*DD*2);
  _Float16* ffw1h = (_Float16*)ws_take(d_ws, off, (size_t)NLAYER*DFF*DD*2);
  _Float16* ffw2h = (_Float16*)ws_take(d_ws, off, (size_t)NLAYER*DD*DFF*2);
  _Float16* linwh = (_Float16*)ws_take(d_ws, off, (size_t)DN*DD*2);
  _Float16* ntlwh = (_Float16*)ws_take(d_ws, off, (size_t)NREL*NHH*DN*DN*2);
  int*      cnt   = (int*)     ws_take(d_ws, off, 4);
  int*      flags = (int*)     ws_take(d_ws, off, (size_t)BB*4);

  auto cvt = [&](const float* s, _Float16* d, int n) {
    f2h_kernel<<<(n + 255)/256, 256, 0, stream>>>(s, d, n);
  };
  // ---- weight conversion (f32 -> f16) ----
  cvt(in_w,  in_wh,  NLAYER*3*DD*DD);
  cvt(out_w, out_wh, NLAYER*DD*DD);
  cvt(ff_w1, ffw1h,  NLAYER*DFF*DD);
  cvt(ff_w2, ffw2h,  NLAYER*DD*DFF);
  cvt(lin_w, linwh,  DN*DD);
  cvt(ntl_w, ntlwh,  NREL*NHH*DN*DN);

  // ---- embedding + positional encoding ----
  embed_kernel<<<(ROWS*DD + 255)/256, 256, 0, stream>>>(
      seq, isvar, iscon, emb, conceptv, variablev, xf, xh);

  // ---- encoder layers ----
  for (int l = 0; l < NLAYER; ++l) {
    // QKV projection (f16 out)
    gemm_wT_kernel<<<dim3(ROWS/16, (3*DD)/64), 32, 0, stream>>>(
        xh, in_wh + (size_t)l*3*DD*DD, in_b + (size_t)l*3*DD,
        nullptr, qkvh, ROWS, 3*DD, DD, 0);
    // fused attention (async V staging + WMMA)
    attention_kernel<<<BB*HH, 256, 0, stream>>>(qkvh, attnh);
    // output projection (f32 out)
    gemm_wT_kernel<<<dim3(ROWS/16, DD/64), 32, 0, stream>>>(
        attnh, out_wh + (size_t)l*DD*DD, out_b + (size_t)l*DD,
        t1f, nullptr, ROWS, DD, DD, 0);
    // residual + LN1
    ln_add_kernel<<<ROWS, 256, 0, stream>>>(xf, t1f,
        ln1_g + (size_t)l*DD, ln1_b + (size_t)l*DD, xf, xh);
    // FFN up (relu, f16 out)
    gemm_wT_kernel<<<dim3(ROWS/16, DFF/64), 32, 0, stream>>>(
        xh, ffw1h + (size_t)l*DFF*DD, ff_b1 + (size_t)l*DFF,
        nullptr, ffh, ROWS, DFF, DD, 1);
    // FFN down (f32 out)
    gemm_wT_kernel<<<dim3(ROWS/16, DD/64), 32, 0, stream>>>(
        ffh, ffw2h + (size_t)l*DD*DFF, ff_b2 + (size_t)l*DD,
        t1f, nullptr, ROWS, DD, DFF, 0);
    // residual + LN2
    ln_add_kernel<<<ROWS, 256, 0, stream>>>(xf, t1f,
        ln2_g + (size_t)l*DD, ln2_b + (size_t)l*DD, xf, xh);
  }

  // ---- final linear to DN ----
  gemm_wT_kernel<<<dim3(ROWS/16, DN/64), 32, 0, stream>>>(
      xh, linwh, lin_b, encf, nullptr, ROWS, DN, DD, 0);

  // ---- gather tokens, segment-sum into node embeddings ----
  node_kernel<<<NN, DN, 0, stream>>>(encf, idx, hf, hh);

  // ---- init logits / counters, then NTL ----
  zero_kernel<<<(MM*NREL + 255)/256, 256, 0, stream>>>(out, MM*NREL);
  init_misc_kernel<<<1, 256, 0, stream>>>(cnt, flags);
  ntl_kernel<<<dim3(MM/16, NREL*NHH), 32, 0, stream>>>(
      hh, hf, srcv, dstv, ntlwh, ntl_v, ntl_bp, ntl_u, out);

  // ---- argmax / acc / emr ----
  score_kernel<<<(MM + 255)/256, 256, 0, stream>>>(out, rel, cnt, flags);
  final_kernel<<<1, 32, 0, stream>>>(cnt, flags, out);
}
